// EnhancedAttentionLayer_41601053229527
// MI455X (gfx1250) — matrix operand order repaired and checked by reading, not compiled, not run on hardware
//
#include <hip/hip_runtime.h>
#include <hip/hip_bf16.h>
#include <stdint.h>

// ---------------- problem constants ----------------
#define B_      8
#define N_      512
#define IN_DIM  768
#define H_      8
#define OUT_DIM 96
#define M_ROWS  (B_ * N_)          // 4096
#define NCOLS   (4 * IN_DIM)       // 3072 : [h1 | h2 | h3 | gate]
#define LEAKY   0.2f

// ---------------- CDNA5 WMMA types ----------------
typedef __attribute__((ext_vector_type(16))) __bf16        v16bf;
typedef __attribute__((ext_vector_type(8)))  float         v8f;
typedef __attribute__((ext_vector_type(4)))  unsigned int  v4u;
typedef __attribute__((ext_vector_type(8)))  unsigned int  v8u;

// round-to-nearest-even f32 -> bf16 (raw bits in ushort)
__device__ __forceinline__ unsigned short f2bf(float f) {
    unsigned int u = __builtin_bit_cast(unsigned int, f);
    u += 0x7FFFu + ((u >> 16) & 1u);
    return (unsigned short)(u >> 16);
}

__device__ __forceinline__ v16bf make_frag(v4u lo, v4u hi) {
    v8u u;
    u.lo = lo;
    u.hi = hi;
    return __builtin_bit_cast(v16bf, u);
}

// ---------------- kernel 0a: feat_in -> bf16 row-major [4096][768] ----------------
__global__ void pack_x_kernel(const float* __restrict__ x,
                              unsigned short* __restrict__ xb, int n) {
    int i = blockIdx.x * blockDim.x + threadIdx.x;
    if (i < n) xb[i] = f2bf(x[i]);
}

// ---------------- kernel 0b: pack W1|W2|W3|H_w^T -> bf16 col-major [3072][768] ----
__global__ void pack_w_kernel(const float* __restrict__ W1,
                              const float* __restrict__ W2,
                              const float* __restrict__ W3,
                              const float* __restrict__ Hw,
                              unsigned short* __restrict__ wp) {
    int i = blockIdx.x * blockDim.x + threadIdx.x;
    if (i >= NCOLS * IN_DIM) return;
    int c = i / IN_DIM;
    int d = i - c * IN_DIM;
    float v;
    if (c < IN_DIM) {
        int h = c / OUT_DIM, e = c % OUT_DIM;
        v = W1[(h * IN_DIM + d) * OUT_DIM + e];
    } else if (c < 2 * IN_DIM) {
        int cc = c - IN_DIM;
        int h = cc / OUT_DIM, e = cc % OUT_DIM;
        v = W2[(h * IN_DIM + d) * OUT_DIM + e];
    } else if (c < 3 * IN_DIM) {
        int cc = c - 2 * IN_DIM;
        int h = cc / OUT_DIM, e = cc % OUT_DIM;
        v = W3[(h * IN_DIM + d) * OUT_DIM + e];
    } else {
        v = Hw[(c - 3 * IN_DIM) * IN_DIM + d];   // gate = feat @ H_w^T
    }
    wp[c * IN_DIM + d] = f2bf(v);
}

// ---------------- kernel 1: fused projection GEMM  [4096,768]x[768,3072] ----------
// one 16x64 output block per wave (4 N-tiles): A fragment reused by 4 WMMAs,
// 2.5 b128 loads per WMMA instead of 4.
__global__ void __launch_bounds__(128)
proj_gemm_kernel(const unsigned short* __restrict__ Xb,
                 const unsigned short* __restrict__ Wp,
                 const float* __restrict__ Hb,
                 float* __restrict__ h1,
                 unsigned short* __restrict__ h2t,
                 unsigned short* __restrict__ h3t,
                 float* __restrict__ gate) {
    const int lane = threadIdx.x & 31;
    const int wave = threadIdx.x >> 5;
    const int tile = blockIdx.x * 4 + wave;      // 16x64 macro-tiles
    const int NT64 = NCOLS / 64;                 // 48
    const int mt   = tile / NT64;
    const int n64  = tile - mt * NT64;
    const int col0 = n64 * 64;
    const int r    = lane & 15;
    const int hi   = lane >> 4;

    // A fragment source: row r of the M-tile, K chunk {koff..+7, koff+16..+23}
    const unsigned short* aptr = Xb + (mt * 16 + r) * IN_DIM + hi * 8;
    // B fragment sources: 4 column tiles, K chunk {koff..+7, koff+16..+23} per lane
    const unsigned short* bptr0 = Wp + (col0 + 0  + r) * IN_DIM + hi * 8;
    const unsigned short* bptr1 = Wp + (col0 + 16 + r) * IN_DIM + hi * 8;
    const unsigned short* bptr2 = Wp + (col0 + 32 + r) * IN_DIM + hi * 8;
    const unsigned short* bptr3 = Wp + (col0 + 48 + r) * IN_DIM + hi * 8;

    v8f acc0 = {}, acc1 = {}, acc2 = {}, acc3 = {};
#pragma unroll 2
    for (int kc = 0; kc < IN_DIM / 32; ++kc) {
        const v4u* pa = (const v4u*)(aptr + kc * 32);
        v16bf a = make_frag(pa[0], pa[2]);       // +0 and +32 bytes

        const v4u* pb0 = (const v4u*)(bptr0 + kc * 32);
        v16bf b0 = make_frag(pb0[0], pb0[2]);
        acc0 = __builtin_amdgcn_wmma_f32_16x16x32_bf16(
            false, a, false, b0, (short)0, acc0, false, false);

        const v4u* pb1 = (const v4u*)(bptr1 + kc * 32);
        v16bf b1 = make_frag(pb1[0], pb1[2]);
        acc1 = __builtin_amdgcn_wmma_f32_16x16x32_bf16(
            false, a, false, b1, (short)0, acc1, false, false);

        const v4u* pb2 = (const v4u*)(bptr2 + kc * 32);
        v16bf b2 = make_frag(pb2[0], pb2[2]);
        acc2 = __builtin_amdgcn_wmma_f32_16x16x32_bf16(
            false, a, false, b2, (short)0, acc2, false, false);

        const v4u* pb3 = (const v4u*)(bptr3 + kc * 32);
        v16bf b3 = make_frag(pb3[0], pb3[2]);
        acc3 = __builtin_amdgcn_wmma_f32_16x16x32_bf16(
            false, a, false, b3, (short)0, acc3, false, false);
    }

    // ---- epilogue: the 64-col group lies entirely in one output region ----
    v8f accs[4] = {acc0, acc1, acc2, acc3};
    const int region = col0 / IN_DIM;            // 0:h1 1:h2t 2:h3t 3:gate (uniform)
    const int cbase  = col0 - region * IN_DIM;

#pragma unroll
    for (int s = 0; s < 4; ++s) {
        const int c = cbase + s * 16 + r;        // column within region, N = lane&15
#pragma unroll
        for (int v = 0; v < 8; ++v) {
            int Mrow = v + hi * 8;               // C layout: M = vgpr + 8*(lane>=16)
            int m  = mt * 16 + Mrow;
            int bb = m >> 9;
            int n  = m & (N_ - 1);
            float val = accs[s][v];
            if (region == 0) {                   // h1, fp32 [B,H,N,96]
                int h = c / OUT_DIM, e = c % OUT_DIM;
                h1[((bb * H_ + h) * N_ + n) * OUT_DIM + e] = val;
            } else if (region == 1) {            // h2 transposed bf16 [B,H,96,512]
                int h = c / OUT_DIM, e = c % OUT_DIM;
                h2t[((bb * H_ + h) * OUT_DIM + e) * N_ + n] = f2bf(val);
            } else if (region == 2) {            // h3 transposed bf16
                int h = c / OUT_DIM, e = c % OUT_DIM;
                h3t[((bb * H_ + h) * OUT_DIM + e) * N_ + n] = f2bf(val);
            } else {                             // gate = sigmoid(acc + H_b)
                float g = 1.0f / (1.0f + __expf(-(val + Hb[c])));
                gate[m * IN_DIM + c] = g;
            }
        }
    }
}

// ---------------- kernel 2: attn_src / attn_dst ------------------------------------
__global__ void attn_vec_kernel(const float* __restrict__ h1,
                                const float* __restrict__ wsrc,
                                const float* __restrict__ wdst,
                                float* __restrict__ asrc,
                                float* __restrict__ adst) {
    int gid = blockIdx.x * blockDim.x + threadIdx.x;   // gid = bh*N + n
    if (gid >= B_ * H_ * N_) return;
    int bh = gid / N_;
    int h  = bh & (H_ - 1);
    const float* row = h1 + (size_t)gid * OUT_DIM;
    const float* ws  = wsrc + h * OUT_DIM;
    const float* wd  = wdst + h * OUT_DIM;
    float s = 0.f, d = 0.f;
#pragma unroll 4
    for (int e = 0; e < OUT_DIM; ++e) {
        float t = tanhf(row[e]);
        s += t * ws[e];
        d += t * wd[e];
    }
    asrc[gid] = s;
    adst[gid] = d;
}

// ---------------- kernel 3: masked dual-softmax attention + output fusion ----------
// one block = 128 threads handles one (b,h) x 16-row tile.
// Phase A: unnormalized exp weights (masked entries exactly 0) -> LDS bf16 tiles
// Phase B: WMMA [16x512]x[512x96] for both attn matrices; epilogue does
//          1/rowsum, diag(adj)*h1, +bias, /3, ELU, sigmoid-gate residual mix.
__global__ void __launch_bounds__(128)
attn_out_kernel(const int* __restrict__ adj,
                const int* __restrict__ smask,
                const float* __restrict__ asrc,
                const float* __restrict__ adst,
                const float* __restrict__ h1,
                const unsigned short* __restrict__ h2t,
                const unsigned short* __restrict__ h3t,
                const float* __restrict__ bias,
                const float* __restrict__ gate,
                const float* __restrict__ feat_in,
                float* __restrict__ out) {
    __shared__ __align__(16) unsigned short A2[16 * N_];
    __shared__ __align__(16) unsigned short A3[16 * N_];
    __shared__ float sum2[16];
    __shared__ float sum3[16];

    const int tid  = threadIdx.x;
    const int bh   = blockIdx.x >> 5;      // (b*8+h)
    const int rt   = blockIdx.x & 31;      // row tile within N
    const int b    = bh >> 3;
    const int h    = bh & 7;
    const int row0 = rt * 16;

    if (tid < 16) { sum2[tid] = 0.f; sum3[tid] = 0.f; }
    __syncthreads();

    // ---- Phase A: logits -> exp -> LDS (bf16), row sums via LDS atomics ----
    {
        const int rl = tid >> 3;           // 16 rows, 8 threads each
        const int lr = tid & 7;
        const int i  = row0 + rl;
        const float si = asrc[bh * N_ + i];
        const int* arow = adj   + ((size_t)b * N_ + i) * N_;
        const int* srow = smask + ((size_t)b * N_ + i) * N_;
        float p2 = 0.f, p3 = 0.f;
        for (int j = lr; j < N_; j += 8) {
            float l = si + adst[bh * N_ + j];
            l = (l >= 0.f) ? l : LEAKY * l;          // leaky_relu
            float ex = __expf(l);                    // bounded: |l| < ~48
            int sm = srow[j];
            int aj = arow[j];
            float e2 = (sm == 1 && i != j) ? ex : 0.f; // inter_clause mask
            float e3 = (sm == 0 && aj == 1) ? ex : 0.f; // outer_clause mask
            A2[rl * N_ + j] = f2bf(e2);
            A3[rl * N_ + j] = f2bf(e3);
            p2 += e2;
            p3 += e3;
        }
        atomicAdd(&sum2[rl], p2);
        atomicAdd(&sum3[rl], p3);
    }
    __syncthreads();

    // ---- Phase B: WMMA over K=512 (16 chunks of 32), 6 e-tiles over 4 waves ----
    const int wave = tid >> 5;
    const int lane = tid & 31;
    const int r    = lane & 15;
    const int hi   = lane >> 4;
    for (int et = wave; et < 6; et += 4) {
        const int e = et * 16 + r;                     // output column
        const unsigned short* b2p = h2t + ((size_t)bh * OUT_DIM + e) * N_ + hi * 8;
        const unsigned short* b3p = h3t + ((size_t)bh * OUT_DIM + e) * N_ + hi * 8;
        v8f c2 = {}, c3 = {};
#pragma unroll 4
        for (int kc = 0; kc < N_ / 32; ++kc) {
            const v4u* pa2 = (const v4u*)&A2[r * N_ + kc * 32 + hi * 8];
            v16bf a2 = make_frag(pa2[0], pa2[2]);
            const v4u* pb2 = (const v4u*)(b2p + kc * 32);
            v16bf f2 = make_frag(pb2[0], pb2[2]);
            c2 = __builtin_amdgcn_wmma_f32_16x16x32_bf16(
                false, a2, false, f2, (short)0, c2, false, false);

            const v4u* pa3 = (const v4u*)&A3[r * N_ + kc * 32 + hi * 8];
            v16bf a3 = make_frag(pa3[0], pa3[2]);
            const v4u* pb3 = (const v4u*)(b3p + kc * 32);
            v16bf f3 = make_frag(pb3[0], pb3[2]);
            c3 = __builtin_amdgcn_wmma_f32_16x16x32_bf16(
                false, a3, false, f3, (short)0, c3, false, false);
        }
        // ---- fused epilogue ----
#pragma unroll
        for (int v = 0; v < 8; ++v) {
            int M = v + hi * 8;
            int n = row0 + M;
            float s2 = sum2[M];
            float s3 = sum3[M];
            float sc2 = (s2 > 0.f) ? 1.0f / s2 : 0.f;  // fully-masked row -> 0
            float sc3 = (s3 > 0.f) ? 1.0f / s3 : 0.f;
            float adjd = (float)adj[((size_t)b * N_ + n) * N_ + n]; // intra=diag(adj)
            float h1v  = h1[((size_t)bh * N_ + n) * OUT_DIM + e];
            float acc  = (adjd * h1v + c2[v] * sc2 + c3[v] * sc3 + bias[e]) * (1.0f / 3.0f);
            float eluv = (acc > 0.f) ? acc : (__expf(acc) - 1.0f);  // ELU
            int gidx   = ((size_t)b * N_ + n) * IN_DIM + h * OUT_DIM + e;
            float g    = gate[gidx];
            out[gidx]  = g * eluv + (1.0f - g) * feat_in[gidx];
        }
    }
}

// ---------------- launcher ----------------
extern "C" void kernel_launch(void* const* d_in, const int* in_sizes, int n_in,
                              void* d_out, int out_size, void* d_ws, size_t ws_size,
                              hipStream_t stream) {
    const float* feat_in = (const float*)d_in[0];
    const int*   adj     = (const int*)d_in[1];
    // d_in[2] = relation (int64) : unused by the reference computation
    const int*   s_mask  = (const int*)d_in[3];
    const float* W1      = (const float*)d_in[4];
    const float* W2      = (const float*)d_in[5];
    const float* W3      = (const float*)d_in[6];
    const float* bias    = (const float*)d_in[7];
    const float* w_src   = (const float*)d_in[8];
    const float* w_dst   = (const float*)d_in[9];
    const float* H_w     = (const float*)d_in[10];
    const float* H_b     = (const float*)d_in[11];
    float* out = (float*)d_out;

    // workspace carve-up (all offsets 256B-aligned by construction)
    char* ws = (char*)d_ws;
    size_t off = 0;
    unsigned short* Xb  = (unsigned short*)(ws + off); off += (size_t)M_ROWS * IN_DIM * 2;   // 6.29 MB
    unsigned short* Wp  = (unsigned short*)(ws + off); off += (size_t)NCOLS * IN_DIM * 2;    // 4.72 MB
    float*          h1  = (float*)(ws + off);          off += (size_t)B_ * H_ * N_ * OUT_DIM * 4; // 12.58 MB
    unsigned short* h2t = (unsigned short*)(ws + off); off += (size_t)B_ * H_ * OUT_DIM * N_ * 2; // 6.29 MB
    unsigned short* h3t = (unsigned short*)(ws + off); off += (size_t)B_ * H_ * OUT_DIM * N_ * 2; // 6.29 MB
    float*          gat = (float*)(ws + off);          off += (size_t)M_ROWS * IN_DIM * 4;   // 12.58 MB
    float*          asr = (float*)(ws + off);          off += (size_t)B_ * H_ * N_ * 4;
    float*          ads = (float*)(ws + off);          off += (size_t)B_ * H_ * N_ * 4;
    (void)ws_size; (void)in_sizes; (void)n_in; (void)out_size;

    // 0) pack inputs to bf16
    {
        int n = M_ROWS * IN_DIM;
        pack_x_kernel<<<(n + 255) / 256, 256, 0, stream>>>(feat_in, Xb, n);
        int m = NCOLS * IN_DIM;
        pack_w_kernel<<<(m + 255) / 256, 256, 0, stream>>>(W1, W2, W3, H_w, Wp);
    }
    // 1) fused projection GEMM (h1 | h2t | h3t | sigmoid gate), 16x64 per wave
    {
        int tiles = (M_ROWS / 16) * (NCOLS / 64);    // 12288 macro-tiles
        proj_gemm_kernel<<<tiles / 4, 128, 0, stream>>>(Xb, Wp, H_b, h1, h2t, h3t, gat);
    }
    // 2) attention source/dest scalars
    {
        int n = B_ * H_ * N_;
        attn_vec_kernel<<<(n + 127) / 128, 128, 0, stream>>>(h1, w_src, w_dst, asr, ads);
    }
    // 3) dual masked softmax + WMMA aggregation + full epilogue
    {
        int blocks = B_ * H_ * (N_ / 16);            // 2048
        attn_out_kernel<<<blocks, 128, 0, stream>>>(adj, s_mask, asr, ads, h1, h2t, h3t,
                                                    bias, gat, feat_in, out);
    }
}
